// MinimalMamba_24773371363450
// MI455X (gfx1250) — compile-verified, matrix operand order checked by reading, not run on hardware
//
#include <hip/hip_runtime.h>
#include <hip/hip_bf16.h>

// ---------------- problem constants (match reference) ----------------
#define DM    1024           // D_MODEL
#define DI    2048           // D_INNER
#define DSN   16             // D_STATE
#define DR    128            // DT_RANK
#define BB    2              // BATCH
#define SQ    2048           // SEQ
#define BS    (BB * SQ)      // 4096 rows for all GEMMs
#define NXD   (DR + 2 * DSN) // 160 = x_proj output width
#define NC    16             // scan chunks
#define CL    (SQ / NC)      // chunk length = 128

typedef __bf16 bf16_t;
typedef bf16_t v16bf __attribute__((ext_vector_type(16)));
typedef float  v8f   __attribute__((ext_vector_type(8)));

__device__ __forceinline__ float sigmoidf_(float x) { return 1.0f / (1.0f + __expf(-x)); }
__device__ __forceinline__ float softplusf_(float x) {
    return (x > 20.0f) ? x : __logf(1.0f + __expf(x));
}

// ------------------------------------------------------------------
// Pack a row-major f32 weight matrix B[K,N] into bf16 WMMA B-operand
// order: tile (nt,kt) -> 32 lanes x 16 elems; elem e of lane holds
// B[kt*32 + (e&7) + (e>>3)*16 + (lane>>4)*8][nt*16 + (lane&15)].
// One thread produces one lane's 16 contiguous bf16 (32 bytes).
// ------------------------------------------------------------------
__global__ __launch_bounds__(256)
void pack_b_kernel(const float* __restrict__ B, int N, int K,
                   bf16_t* __restrict__ out, int total_tiles)
{
    const int idx  = blockIdx.x * 256 + threadIdx.x;
    const int lane = idx & 31;
    const int tile = idx >> 5;                 // tile = nt * ktiles + kt
    if (tile >= total_tiles) return;

    const int ktiles = K >> 5;
    const int kt = tile % ktiles;
    const int nt = tile / ktiles;
    const int lo   = lane & 15;
    const int half = (lane >> 4) & 1;

    v16bf v;
    #pragma unroll
    for (int e = 0; e < 16; ++e) {
        const int kk = kt * 32 + (e & 7) + ((e >> 3) << 4) + half * 8;
        v[e] = (bf16_t)B[(size_t)kk * N + nt * 16 + lo];
    }
    *((v16bf*)out + (size_t)tile * 32 + lane) = v;
}

// ------------------------------------------------------------------
// GEMM: C[M,N] = A[M,K](f32) * Bpacked[K,N](bf16 operand layout)
// Block = 128 threads (4 waves). Wave w -> 32x32 tile at
// (m0 = (blockIdx.y*4+w)*32, n0 = blockIdx.x*32): 2 A x 2 B operands,
// 4 x v_wmma_f32_16x16x32_bf16 per 32-K step.
// Requires M % 128 == 0, N % 32 == 0, K % 32 == 0 (true for all calls).
// EPI == 1 : softplus(acc + bias[n])   (dt_proj epilogue)
// ------------------------------------------------------------------
template <int EPI>
__global__ __launch_bounds__(128)
void wmma_gemm_packed(const float* __restrict__ A, int lda,
                      const bf16_t* __restrict__ Bp,
                      float* __restrict__ C, int ldc,
                      int K, const float* __restrict__ bias)
{
    const int lane = threadIdx.x & 31;
    const int wave = threadIdx.x >> 5;
    const int lo   = lane & 15;
    const int half = (lane >> 4) & 1;

    const int m0     = (blockIdx.y * 4 + wave) * 32;
    const int n0     = blockIdx.x * 32;
    const int ktiles = K >> 5;
    const int nt0    = blockIdx.x * 2;

    v8f acc00 = {}, acc01 = {}, acc10 = {}, acc11 = {};

    const float*  arow0 = A + (size_t)(m0 + lo) * lda;
    const float*  arow1 = A + (size_t)(m0 + 16 + lo) * lda;
    const v16bf*  bp0   = (const v16bf*)Bp + (size_t)(nt0 + 0) * ktiles * 32 + lane;
    const v16bf*  bp1   = (const v16bf*)Bp + (size_t)(nt0 + 1) * ktiles * 32 + lane;

    for (int kt = 0; kt < ktiles; ++kt) {
        const int k0 = kt * 32;
        // prefetch next A segments (gfx1250 global_prefetch_b8)
        __builtin_prefetch(arow0 + k0 + 64, 0, 1);
        __builtin_prefetch(arow1 + k0 + 64, 0, 1);

        // A operands: elem e -> K = k0 + (e&7) + (e>>3)*16 + half*8
        v16bf a0, a1;
        {
            const float* ap = arow0 + k0 + half * 8;
            float4 q0 = *(const float4*)(ap + 0);
            float4 q1 = *(const float4*)(ap + 4);
            float4 q2 = *(const float4*)(ap + 16);
            float4 q3 = *(const float4*)(ap + 20);
            a0[0]=(bf16_t)q0.x; a0[1]=(bf16_t)q0.y; a0[2]=(bf16_t)q0.z; a0[3]=(bf16_t)q0.w;
            a0[4]=(bf16_t)q1.x; a0[5]=(bf16_t)q1.y; a0[6]=(bf16_t)q1.z; a0[7]=(bf16_t)q1.w;
            a0[8]=(bf16_t)q2.x; a0[9]=(bf16_t)q2.y; a0[10]=(bf16_t)q2.z; a0[11]=(bf16_t)q2.w;
            a0[12]=(bf16_t)q3.x; a0[13]=(bf16_t)q3.y; a0[14]=(bf16_t)q3.z; a0[15]=(bf16_t)q3.w;
        }
        {
            const float* ap = arow1 + k0 + half * 8;
            float4 q0 = *(const float4*)(ap + 0);
            float4 q1 = *(const float4*)(ap + 4);
            float4 q2 = *(const float4*)(ap + 16);
            float4 q3 = *(const float4*)(ap + 20);
            a1[0]=(bf16_t)q0.x; a1[1]=(bf16_t)q0.y; a1[2]=(bf16_t)q0.z; a1[3]=(bf16_t)q0.w;
            a1[4]=(bf16_t)q1.x; a1[5]=(bf16_t)q1.y; a1[6]=(bf16_t)q1.z; a1[7]=(bf16_t)q1.w;
            a1[8]=(bf16_t)q2.x; a1[9]=(bf16_t)q2.y; a1[10]=(bf16_t)q2.z; a1[11]=(bf16_t)q2.w;
            a1[12]=(bf16_t)q3.x; a1[13]=(bf16_t)q3.y; a1[14]=(bf16_t)q3.z; a1[15]=(bf16_t)q3.w;
        }
        // B operands: contiguous 32B per lane from the packed buffer
        v16bf b0 = bp0[(size_t)kt * 32];
        v16bf b1 = bp1[(size_t)kt * 32];

        acc00 = __builtin_amdgcn_wmma_f32_16x16x32_bf16(false, a0, false, b0, (short)0, acc00, false, false);
        acc01 = __builtin_amdgcn_wmma_f32_16x16x32_bf16(false, a0, false, b1, (short)0, acc01, false, false);
        acc10 = __builtin_amdgcn_wmma_f32_16x16x32_bf16(false, a1, false, b0, (short)0, acc10, false, false);
        acc11 = __builtin_amdgcn_wmma_f32_16x16x32_bf16(false, a1, false, b1, (short)0, acc11, false, false);
    }

    // store: D elem i -> (M = m0 + i + 8*half [+16], N = n0 + lo [+16])
    #pragma unroll
    for (int i = 0; i < 8; ++i) {
        const size_t r0 = (size_t)(m0 + i + 8 * half);
        const size_t r1 = r0 + 16;
        const int c0 = n0 + lo;
        const int c1 = c0 + 16;
        float v00 = acc00[i], v01 = acc01[i], v10 = acc10[i], v11 = acc11[i];
        if (EPI == 1) {
            v00 = softplusf_(v00 + bias[c0]);
            v01 = softplusf_(v01 + bias[c1]);
            v10 = softplusf_(v10 + bias[c0]);
            v11 = softplusf_(v11 + bias[c1]);
        }
        C[r0 * ldc + c0] = v00;
        C[r0 * ldc + c1] = v01;
        C[r1 * ldc + c0] = v10;
        C[r1 * ldc + c1] = v11;
    }
}

// ------------------------------------------------------------------
// Depthwise causal conv1d (k=4) + SiLU on the xb half of xz.
// ------------------------------------------------------------------
__global__ __launch_bounds__(256)
void conv_silu_kernel(const float* __restrict__ xz,
                      const float* __restrict__ cw,   // [4][1][DI]
                      const float* __restrict__ cb,   // [DI]
                      float* __restrict__ xbc)        // [BS, DI]
{
    const int idx = blockIdx.x * 256 + threadIdx.x;   // < BS*DI
    const int d = idx % DI;
    const int t = (idx / DI) % SQ;
    const int b = idx / (DI * SQ);

    float acc = cb[d];
    #pragma unroll
    for (int k = 0; k < 4; ++k) {
        const int ts = t + k - 3;
        if (ts >= 0)
            acc += cw[k * DI + d] * xz[(size_t)(b * SQ + ts) * (2 * DI) + d];
    }
    acc = acc * sigmoidf_(acc);
    xbc[(size_t)(b * SQ + t) * DI + d] = acc;
}

// ------------------------------------------------------------------
// Chunked selective scan. Diagonal SSM => chunk decay = exp(a[n]*sum_dt).
// Pass 1: per-(b,d,chunk) local scan from h=0 -> h_end[16], sum_dt.
// Pass 2: per-(b,d) sequential combine over 16 chunks -> h_start[16].
// Pass 3: per-(b,d,chunk) re-scan from h_start, write fused output.
// ------------------------------------------------------------------
__global__ __launch_bounds__(256)
void scan_local_kernel(const float* __restrict__ xbc,
                       const float* __restrict__ xdbl,
                       const float* __restrict__ dtbuf,
                       const float* __restrict__ A_log,
                       float* __restrict__ HEND,    // [BB*DI*NC, 16]
                       float* __restrict__ SUMDT)   // [BB*DI*NC]
{
    const int idx = blockIdx.x * 256 + threadIdx.x;   // < BB*DI*NC
    const int d = idx % DI;
    const int c = (idx / DI) % NC;
    const int b = idx / (DI * NC);

    float a[DSN], h[DSN];
    #pragma unroll
    for (int n = 0; n < DSN; ++n) { a[n] = -__expf(A_log[d * DSN + n]); h[n] = 0.0f; }

    float sumdt = 0.0f;
    const size_t rbase = (size_t)b * SQ + (size_t)c * CL;
    for (int tt = 0; tt < CL; ++tt) {
        const size_t r = rbase + tt;
        const float dt = dtbuf[r * DI + d];
        const float xv = xbc[r * DI + d];
        const float dx = dt * xv;
        sumdt += dt;
        #pragma unroll
        for (int n = 0; n < DSN; ++n) {
            const float Bm = xdbl[r * NXD + DR + n];
            h[n] = __expf(dt * a[n]) * h[n] + dx * Bm;
        }
    }
    const size_t o = (size_t)(b * DI + d) * NC + c;
    #pragma unroll
    for (int n = 0; n < DSN; ++n) HEND[o * DSN + n] = h[n];
    SUMDT[o] = sumdt;
}

__global__ __launch_bounds__(256)
void scan_combine_kernel(const float* __restrict__ A_log,
                         const float* __restrict__ HEND,
                         const float* __restrict__ SUMDT,
                         float* __restrict__ HSTART)  // [BB*DI*NC, 16]
{
    const int idx = blockIdx.x * 256 + threadIdx.x;   // < BB*DI
    const int d = idx % DI;
    const int b = idx / DI;

    float a[DSN], h[DSN];
    #pragma unroll
    for (int n = 0; n < DSN; ++n) { a[n] = -__expf(A_log[d * DSN + n]); h[n] = 0.0f; }

    const size_t base = (size_t)(b * DI + d) * NC;
    for (int c = 0; c < NC; ++c) {
        #pragma unroll
        for (int n = 0; n < DSN; ++n) HSTART[(base + c) * DSN + n] = h[n];
        const float sd = SUMDT[base + c];
        #pragma unroll
        for (int n = 0; n < DSN; ++n)
            h[n] = __expf(sd * a[n]) * h[n] + HEND[(base + c) * DSN + n];
    }
}

__global__ __launch_bounds__(256)
void scan_final_kernel(const float* __restrict__ xbc,
                       const float* __restrict__ xdbl,
                       const float* __restrict__ dtbuf,
                       const float* __restrict__ xz,
                       const float* __restrict__ A_log,
                       const float* __restrict__ Dw,
                       const float* __restrict__ HSTART,
                       float* __restrict__ ymul)     // [BS, DI]
{
    const int idx = blockIdx.x * 256 + threadIdx.x;   // < BB*DI*NC
    const int d = idx % DI;
    const int c = (idx / DI) % NC;
    const int b = idx / (DI * NC);

    float a[DSN], h[DSN];
    const size_t o = (size_t)(b * DI + d) * NC + c;
    #pragma unroll
    for (int n = 0; n < DSN; ++n) {
        a[n] = -__expf(A_log[d * DSN + n]);
        h[n] = HSTART[o * DSN + n];
    }
    const float Dd = Dw[d];

    const size_t rbase = (size_t)b * SQ + (size_t)c * CL;
    for (int tt = 0; tt < CL; ++tt) {
        const size_t r = rbase + tt;
        const float dt = dtbuf[r * DI + d];
        const float xv = xbc[r * DI + d];
        const float dx = dt * xv;
        float y = 0.0f;
        #pragma unroll
        for (int n = 0; n < DSN; ++n) {
            const float Bm = xdbl[r * NXD + DR + n];
            const float Cm = xdbl[r * NXD + DR + DSN + n];
            h[n] = __expf(dt * a[n]) * h[n] + dx * Bm;
            y += h[n] * Cm;
        }
        y += xv * Dd;
        const float z = xz[r * (2 * DI) + DI + d];
        ymul[r * DI + d] = y * (z * sigmoidf_(z));
    }
}

// ------------------------------------------------------------------
extern "C" void kernel_launch(void* const* d_in, const int* in_sizes, int n_in,
                              void* d_out, int out_size, void* d_ws, size_t ws_size,
                              hipStream_t stream)
{
    (void)in_sizes; (void)n_in; (void)out_size; (void)ws_size;

    const float* x        = (const float*)d_in[0]; // [BB,SQ,DM]
    const float* in_proj  = (const float*)d_in[1]; // [DM, 2*DI]
    const float* conv_w   = (const float*)d_in[2]; // [4,1,DI]
    const float* conv_b   = (const float*)d_in[3]; // [DI]
    const float* x_proj   = (const float*)d_in[4]; // [DI, NXD]
    const float* dt_proj  = (const float*)d_in[5]; // [DR, DI]
    const float* dt_b     = (const float*)d_in[6]; // [DI]
    const float* A_log    = (const float*)d_in[7]; // [DI, DSN]
    const float* Dw       = (const float*)d_in[8]; // [DI]
    const float* out_proj = (const float*)d_in[9]; // [DI, DM]
    float* out = (float*)d_out;                    // [BB,SQ,DM]

    // ---- workspace layout (all offsets 256B aligned) ----
    char* ws = (char*)d_ws;
    size_t off = 0;
    auto alloc = [&](size_t bytes) {
        char* p = ws + off;
        off += (bytes + 255) & ~(size_t)255;
        return p;
    };
    float*  XZ     = (float*)alloc((size_t)BS * (2 * DI) * 4);   // 64 MB
    float*  XBC    = (float*)alloc((size_t)BS * DI * 4);         // 32 MB
    float*  XDBL   = (float*)alloc((size_t)BS * NXD * 4);        // 2.6 MB
    float*  DT     = (float*)alloc((size_t)BS * DI * 4);         // 32 MB
    float*  YMUL   = (float*)alloc((size_t)BS * DI * 4);         // 32 MB
    bf16_t* WP_IN  = (bf16_t*)alloc((size_t)DM * (2 * DI) * 2);  // 8 MB
    bf16_t* WP_XP  = (bf16_t*)alloc((size_t)DI * NXD * 2);       // 640 KB
    bf16_t* WP_DT  = (bf16_t*)alloc((size_t)DR * DI * 2);        // 512 KB
    bf16_t* WP_OUT = (bf16_t*)alloc((size_t)DI * DM * 2);        // 4 MB
    float*  HEND   = (float*)alloc((size_t)BB * DI * NC * DSN * 4); // 4 MB
    float*  HSTART = (float*)alloc((size_t)BB * DI * NC * DSN * 4); // 4 MB
    float*  SUMDT  = (float*)alloc((size_t)BB * DI * NC * 4);       // 256 KB

    // ---- pack weights to bf16 WMMA B-operand layout ----
    {
        int tiles;
        tiles = ((2 * DI) / 16) * (DM / 32);   // in_proj: 256*32
        pack_b_kernel<<<(tiles * 32 + 255) / 256, 256, 0, stream>>>(in_proj, 2 * DI, DM, WP_IN, tiles);
        tiles = (NXD / 16) * (DI / 32);        // x_proj: 10*64
        pack_b_kernel<<<(tiles * 32 + 255) / 256, 256, 0, stream>>>(x_proj, NXD, DI, WP_XP, tiles);
        tiles = (DI / 16) * (DR / 32);         // dt_proj: 128*4
        pack_b_kernel<<<(tiles * 32 + 255) / 256, 256, 0, stream>>>(dt_proj, DI, DR, WP_DT, tiles);
        tiles = (DM / 16) * (DI / 32);         // out_proj: 64*64
        pack_b_kernel<<<(tiles * 32 + 255) / 256, 256, 0, stream>>>(out_proj, DM, DI, WP_OUT, tiles);
    }

    // 1) xz = x @ in_proj_w                      (4096 x 1024 x 4096)
    wmma_gemm_packed<0><<<dim3((2 * DI) / 32, BS / 128), 128, 0, stream>>>(
        x, DM, WP_IN, XZ, 2 * DI, DM, nullptr);

    // 2) xbc = silu(depthwise_causal_conv(xb) + conv_b)
    conv_silu_kernel<<<(BS * DI) / 256, 256, 0, stream>>>(XZ, conv_w, conv_b, XBC);

    // 3) x_dbl = xbc @ x_proj_w                  (4096 x 2048 x 160)
    wmma_gemm_packed<0><<<dim3(NXD / 32, BS / 128), 128, 0, stream>>>(
        XBC, DI, WP_XP, XDBL, NXD, DI, nullptr);

    // 4) dt = softplus(x_dbl[:, :DR] @ dt_proj_w + dt_proj_b)  (4096 x 128 x 2048)
    wmma_gemm_packed<1><<<dim3(DI / 32, BS / 128), 128, 0, stream>>>(
        XDBL, NXD, WP_DT, DT, DI, DR, dt_b);

    // 5) chunked selective scan + fused (y + x*D) * silu(z)
    scan_local_kernel<<<(BB * DI * NC) / 256, 256, 0, stream>>>(XBC, XDBL, DT, A_log, HEND, SUMDT);
    scan_combine_kernel<<<(BB * DI) / 256, 256, 0, stream>>>(A_log, HEND, SUMDT, HSTART);
    scan_final_kernel<<<(BB * DI * NC) / 256, 256, 0, stream>>>(XBC, XDBL, DT, XZ, A_log, Dw, HSTART, YMUL);

    // 6) out = ymul @ out_proj_w                 (4096 x 2048 x 1024)
    wmma_gemm_packed<0><<<dim3(DM / 32, BS / 128), 128, 0, stream>>>(
        YMUL, DI, WP_OUT, out, DM, DI, nullptr);
}
